// PLANAttention_22986664968484
// MI455X (gfx1250) — matrix operand order repaired and checked by reading, not compiled, not run on hardware
//
#include <hip/hip_runtime.h>
#include <stdint.h>

// ---------------------------------------------------------------------------
// PLAN attention for MI455X (gfx1250), wave32, bf16 WMMA 16x16x32 pipeline.
// All GEMM operands staged as bf16 in fragment-friendly layouts so inner
// loops are pure vector-load + v_wmma (no per-element conversion VALU).
// ---------------------------------------------------------------------------

typedef __attribute__((ext_vector_type(16))) __bf16    bf16x16;
typedef __attribute__((ext_vector_type(8)))  float     f32x8;
typedef __attribute__((ext_vector_type(8)))  unsigned  u32x8;
typedef unsigned short ush;

#define WSZ 32

// Problem constants
#define Bsz 2
#define Cch 256
#define Hh 40
#define Ww 40
#define HBp 80
#define WBp 80
#define NHh 8
#define Dd 32
#define Bn 16
#define NPIX_T 1600
#define NPIX_B 6400
#define UNITS 25600

// ---- bf16 helpers ---------------------------------------------------------
__device__ __forceinline__ unsigned bf_rne_bits(float f) {
  unsigned u = __builtin_bit_cast(unsigned, f);
  return (u + 0x7FFFu + ((u >> 16) & 1u)) >> 16;           // round-nearest-even
}
__device__ __forceinline__ unsigned pack2_rne(float a, float b) {
  return bf_rne_bits(a) | (bf_rne_bits(b) << 16);
}
// truncating pack of two f32 high-halves into one dword: single v_perm_b32
__device__ __forceinline__ unsigned pack2_trunc(float a, float b) {
  return __builtin_amdgcn_perm(__builtin_bit_cast(unsigned, b),
                               __builtin_bit_cast(unsigned, a), 0x07060302u);
}

// K-index inside a 16x32 bf16 A fragment for element i (0..15), lane group g.
// contiguous A layout: element i of group g -> K{16g+i}
__device__ __forceinline__ int a_kidx(int i, int g) {
  int v = i >> 1;
  return 16 * g + 2 * v + (i & 1);
}

// ---------------------------------------------------------------------------
// zero the f32 fold accumulator (graph replays -> must re-zero every call)
// ---------------------------------------------------------------------------
__global__ void zero_kernel(float* __restrict__ p, unsigned n) {
  unsigned i = blockIdx.x * blockDim.x + threadIdx.x;
  unsigned stride = gridDim.x * blockDim.x;
  for (; i < n; i += stride) p[i] = 0.0f;
}

// ---------------------------------------------------------------------------
// Weight prep: 8 matrices (256x256 f32) -> fragment-ready bf16.
// Linear index tid = ((((mat*16+mt)*8+kk)*2+g)*16+ln); each thread emits the
// 16 a_kidx-ordered elements for (A-row = mt*16+ln, K-chunk kk, group g)
// as 32 contiguous bytes -> GEMM A-load is one bf16x16 vector load.
// ---------------------------------------------------------------------------
__global__ void prep_w(const float* __restrict__ W0, const float* __restrict__ W1,
                       const float* __restrict__ W2, const float* __restrict__ W3,
                       const float* __restrict__ W4, const float* __restrict__ W5,
                       const float* __restrict__ W6, const float* __restrict__ W7,
                       unsigned* __restrict__ WAu) {
  int tid = blockIdx.x * blockDim.x + threadIdx.x;         // 32768 total
  int ln = tid & 15, g = (tid >> 4) & 1, kk = (tid >> 5) & 7;
  int mt = (tid >> 8) & 15, mat = tid >> 12;
  const float* W = (mat == 0) ? W0 : (mat == 1) ? W1 : (mat == 2) ? W2 :
                   (mat == 3) ? W3 : (mat == 4) ? W4 : (mat == 5) ? W5 :
                   (mat == 6) ? W6 : W7;
  const float* row = W + (size_t)(mt * 16 + ln) * Cch + kk * 32;
  unsigned* dst = WAu + (size_t)tid * 8;
#pragma unroll
  for (int v = 0; v < 8; ++v) {
    int k0 = a_kidx(2 * v, g);                             // a_kidx(2v+1)=k0+1
    dst[v] = pack2_rne(row[k0], row[k0 + 1]);
  }
}

// ---------------------------------------------------------------------------
// Activation prep: X (Bsz,256,npix) f32 -> Xt (2*npix, 256) bf16 (c-major).
// Reads coalesced across threads (thread = pixel); one pass over the input.
// ---------------------------------------------------------------------------
__global__ void prep_x(const float* __restrict__ X, unsigned* __restrict__ Xtu,
                       int npix) {
  int pix = blockIdx.x * blockDim.x + threadIdx.x;
  if (pix >= 2 * npix) return;
  int b = pix / npix, hw = pix % npix;
  const float* src = X + (size_t)b * Cch * npix + hw;
  unsigned* dst = Xtu + (size_t)pix * (Cch / 2);
#pragma unroll 4
  for (int c = 0; c < Cch; c += 2)
    dst[c >> 1] = pack2_rne(src[(size_t)c * npix], src[(size_t)(c + 1) * npix]);
}

// ---------------------------------------------------------------------------
// Projection GEMM (K,Q,V): one wave computes a 16(M) x 64(N) tile.
// Per K-chunk: 1 A vector-load (reused 4x) + 4 B vector-loads + 4 WMMAs.
// Output packed bf16 -> P[proj][bn=b*8+head][pix][d], one b128 store/tile-col.
// ---------------------------------------------------------------------------
__global__ void __launch_bounds__(128)
proj_gemm(const ush* __restrict__ Xt, const ush* __restrict__ WA,
          unsigned* __restrict__ P, int npix, int ntN4, int matBase) {
  int wid  = (blockIdx.x * blockDim.x + threadIdx.x) >> 5;
  int lane = threadIdx.x & 31, g = lane >> 4, ln = lane & 15;
  int tilesPerProj = 16 * ntN4;
  int proj = wid / tilesPerProj;
  if (proj >= 3) return;                                   // wave-uniform
  int t  = wid % tilesPerProj;
  int mt = t / ntN4, nq = t % ntN4;
  int mat = matBase + proj;
  int pix0 = nq * 64 + ln;

  f32x8 acc[4] = {};
  for (int kk = 0; kk < 8; ++kk) {
    bf16x16 a = *(const bf16x16*)(WA +
        (size_t)(((((mat * 16 + mt) * 8 + kk) * 2 + g) * 16 + ln)) * 16);
#pragma unroll
    for (int q = 0; q < 4; ++q) {
      bf16x16 bb = *(const bf16x16*)(Xt +
          (size_t)(pix0 + q * 16) * Cch + kk * 32 + 16 * g);
      acc[q] = __builtin_amdgcn_wmma_f32_16x16x32_bf16(false, a, false, bb,
                                                       (short)0, acc[q], false, false);
    }
  }
  int d0 = (mt * 16 + 8 * g) & 31;                         // 8 consecutive d
  int head = (mt * 16 + 8 * g) >> 5;
#pragma unroll
  for (int q = 0; q < 4; ++q) {
    int pix = pix0 + q * 16;
    int b = pix / npix, hw = pix % npix;
    unsigned* dst = P + ((((size_t)proj * Bn + b * NHh + head) * npix + hw) * Dd + d0) / 2;
#pragma unroll
    for (int v = 0; v < 4; ++v)
      dst[v] = pack2_rne(acc[q][2 * v], acc[q][2 * v + 1]); // merges to b128
  }
}

// ---------------------------------------------------------------------------
// Attention element gather (bf16): reproduces the reference's scrambled
// reshape: row 0 = top row; row m>0: f=(m-1)*32+e, dd=f/25, tap t=f%25.
// ---------------------------------------------------------------------------
__device__ __forceinline__ unsigned mat_bf(const ush* __restrict__ topPp,
                                           const ush* __restrict__ botPp,
                                           int bn, int hw, int h, int w,
                                           int m, int e) {
  if (m == 0) return topPp[((size_t)bn * NPIX_T + hw) * Dd + e];
  if (m > 25) return 0u;
  int f  = (m - 1) * 32 + e;
  int dd = f / 25, t = f - dd * 25;
  int ki = t / 5,  kj = t - ki * 5;
  int hb = 2 * h - 2 + ki, wb = 2 * w - 2 + kj;
  if ((unsigned)hb >= (unsigned)HBp || (unsigned)wb >= (unsigned)WBp) return 0u;
  return botPp[((size_t)bn * NPIX_B + (size_t)(hb * WBp + wb)) * Dd + dd];
}

// ---------------------------------------------------------------------------
// Attention: one wave per (bn,h,w) unit. S = K Q^T / sqrt(32) (4 WMMAs on
// padded 32x32), staged in per-wave LDS (f32, stride 33), then O = S V
// (4 WMMAs).  Row 0 -> topRb (bf16); rows 1..25 -> fold scatter (f32 atomics).
// ---------------------------------------------------------------------------
__global__ void __launch_bounds__(128)
attn_kernel(const ush* __restrict__ topP, const ush* __restrict__ botP,
            ush* __restrict__ topRb, float* __restrict__ botR) {
  __shared__ float sS[4][32 * 33];
  int wave = threadIdx.x >> 5;
  int lane = threadIdx.x & 31, g = lane >> 4, ln = lane & 15;
  int unit = blockIdx.x * 4 + wave;                        // grid exact
  int bn = unit / NPIX_T, hw = unit % NPIX_T;
  int h = hw / Ww, w = hw % Ww;

  const ush* topK = topP;
  const ush* topQ = topP + (size_t)Bn * NPIX_T * Dd;
  const ush* topV = topP + (size_t)2 * Bn * NPIX_T * Dd;
  const ush* botK = botP;
  const ush* botQ = botP + (size_t)Bn * NPIX_B * Dd;
  const ush* botV = botP + (size_t)2 * Bn * NPIX_B * Dd;
  float* S = sS[wave];

  // ---- S = K Q^T -------------------------------------------------------
  bf16x16 aK[2], bQ[2];
#pragma unroll
  for (int Mt = 0; Mt < 2; ++Mt) {
    int m = Mt * 16 + ln;
    u32x8 au;
#pragma unroll
    for (int v = 0; v < 8; ++v) {
      int k0 = a_kidx(2 * v, g);
      unsigned lo = mat_bf(topK, botK, bn, hw, h, w, m, k0);
      unsigned hi = mat_bf(topK, botK, bn, hw, h, w, m, k0 + 1);
      au[v] = lo | (hi << 16);
    }
    aK[Mt] = __builtin_bit_cast(bf16x16, au);
  }
#pragma unroll
  for (int Nt = 0; Nt < 2; ++Nt) {
    int j = Nt * 16 + ln;
    u32x8 au;
#pragma unroll
    for (int v = 0; v < 8; ++v) {
      int e0 = 16 * g + 2 * v;
      unsigned lo = mat_bf(topQ, botQ, bn, hw, h, w, j, e0);
      unsigned hi = mat_bf(topQ, botQ, bn, hw, h, w, j, e0 + 1);
      au[v] = lo | (hi << 16);
    }
    bQ[Nt] = __builtin_bit_cast(bf16x16, au);
  }
  const float scale = 0.17677669529663687f;                // 1/sqrt(DK)
#pragma unroll
  for (int Mt = 0; Mt < 2; ++Mt)
#pragma unroll
    for (int Nt = 0; Nt < 2; ++Nt) {
      f32x8 s = {};
      s = __builtin_amdgcn_wmma_f32_16x16x32_bf16(false, aK[Mt], false, bQ[Nt],
                                                  (short)0, s, false, false);
#pragma unroll
      for (int r = 0; r < 8; ++r)
        S[(Mt * 16 + r + 8 * g) * 33 + (Nt * 16 + ln)] = s[r] * scale;
    }

  // same-wave LDS write->read: DS ops in order, compiler inserts DScnt waits

  // ---- O = S V ---------------------------------------------------------
  bf16x16 aS[2], bV[2];
#pragma unroll
  for (int Mt = 0; Mt < 2; ++Mt) {
    int m = Mt * 16 + ln;
    u32x8 au;
#pragma unroll
    for (int v = 0; v < 8; ++v) {
      int k0 = a_kidx(2 * v, g);
      au[v] = pack2_trunc(S[m * 33 + k0], S[m * 33 + k0 + 1]); // 1 v_perm/pair
    }
    aS[Mt] = __builtin_bit_cast(bf16x16, au);
  }
#pragma unroll
  for (int Nt = 0; Nt < 2; ++Nt) {
    int vcol = Nt * 16 + ln;
    u32x8 au;
#pragma unroll
    for (int v = 0; v < 8; ++v) {
      int j0 = 16 * g + 2 * v;
      unsigned lo = mat_bf(topV, botV, bn, hw, h, w, j0, vcol);
      unsigned hi = mat_bf(topV, botV, bn, hw, h, w, j0 + 1, vcol);
      au[v] = lo | (hi << 16);
    }
    bV[Nt] = __builtin_bit_cast(bf16x16, au);
  }
#pragma unroll
  for (int Mt = 0; Mt < 2; ++Mt)
#pragma unroll
    for (int Nt = 0; Nt < 2; ++Nt) {
      f32x8 o = {};
      o = __builtin_amdgcn_wmma_f32_16x16x32_bf16(false, aS[Mt], false, bV[Nt],
                                                  (short)0, o, false, false);
#pragma unroll
      for (int r = 0; r < 8; ++r) {
        int irow = Mt * 16 + r + 8 * g;
        int vcol = Nt * 16 + ln;
        float val = o[r];
        if (irow == 0) {
          topRb[((size_t)bn * NPIX_T + hw) * Dd + vcol] = (ush)bf_rne_bits(val);
        } else if (irow <= 25) {
          int f  = (irow - 1) * 32 + vcol;                 // inverse permutation
          int dd = f / 25, t = f - dd * 25;
          int ki = t / 5,  kj = t - ki * 5;
          int hb = 2 * h - 2 + ki, wb = 2 * w - 2 + kj;
          if ((unsigned)hb < (unsigned)HBp && (unsigned)wb < (unsigned)WBp) {
            __hip_atomic_fetch_add(
                &botR[((size_t)bn * NPIX_B + (size_t)(hb * WBp + wb)) * Dd + dd],
                val, __ATOMIC_RELAXED, __HIP_MEMORY_SCOPE_AGENT);
          }
        }
      }
    }
}

// ---------------------------------------------------------------------------
// Convert fold accumulator f32 -> bf16 so fuse GEMM runs the lean path.
// ---------------------------------------------------------------------------
__global__ void conv_bf(const float* __restrict__ src, unsigned* __restrict__ dst,
                        unsigned n2) {
  unsigned i = blockIdx.x * blockDim.x + threadIdx.x;
  if (i < n2) dst[i] = pack2_rne(src[2 * i], src[2 * i + 1]);
}

// ---------------------------------------------------------------------------
// Fuse GEMM: Y(b,o,pix) = sum_c Wf[o][c] * R[b*8 + c/32][pix][c%32]
// R is bf16 [bn][pix][32]; B fragment (16 consecutive c within one head) is
// a single bf16x16 vector load.  16(M) x 64(N) per wave.
// ---------------------------------------------------------------------------
__global__ void __launch_bounds__(128)
fuse_gemm(const ush* __restrict__ Rb, const ush* __restrict__ WA,
          float* __restrict__ Y, int npix, int ntN4, int mat) {
  int wid  = (blockIdx.x * blockDim.x + threadIdx.x) >> 5;
  int lane = threadIdx.x & 31, g = lane >> 4, ln = lane & 15;
  int mt = wid / ntN4, nq = wid % ntN4;
  if (mt >= 16) return;                                    // wave-uniform
  int pix0 = nq * 64 + ln;

  f32x8 acc[4] = {};
  for (int kk = 0; kk < 8; ++kk) {
    bf16x16 a = *(const bf16x16*)(WA +
        (size_t)(((((mat * 16 + mt) * 8 + kk) * 2 + g) * 16 + ln)) * 16);
    int c0 = kk * 32 + 16 * g;
    int head = c0 >> 5, dd0 = c0 & 31;
#pragma unroll
    for (int q = 0; q < 4; ++q) {
      int pix = pix0 + q * 16;
      int b = pix / npix, hw = pix % npix;
      bf16x16 bb = *(const bf16x16*)(Rb +
          (((size_t)b * NHh + head) * npix + hw) * Dd + dd0);
      acc[q] = __builtin_amdgcn_wmma_f32_16x16x32_bf16(false, a, false, bb,
                                                       (short)0, acc[q], false, false);
    }
  }
#pragma unroll
  for (int q = 0; q < 4; ++q) {
    int pix = pix0 + q * 16;
    int b = pix / npix, hw = pix % npix;
#pragma unroll
    for (int r = 0; r < 8; ++r) {
      int oo = mt * 16 + r + 8 * g;
      Y[((size_t)b * Cch + oo) * npix + hw] = acc[q][r];
    }
  }
}

// ---------------------------------------------------------------------------
extern "C" void kernel_launch(void* const* d_in, const int* in_sizes, int n_in,
                              void* d_out, int out_size, void* d_ws, size_t ws_size,
                              hipStream_t stream) {
  const float* top  = (const float*)d_in[0];
  const float* bot  = (const float*)d_in[1];
  const float* w_tk = (const float*)d_in[2];
  const float* w_tq = (const float*)d_in[3];
  const float* w_tv = (const float*)d_in[4];
  const float* w_bk = (const float*)d_in[5];
  const float* w_bq = (const float*)d_in[6];
  const float* w_bv = (const float*)d_in[7];
  const float* w_tf = (const float*)d_in[8];
  const float* w_bf = (const float*)d_in[9];
  float* out = (float*)d_out;

  // workspace carve (bf16 region first, 16B-aligned f32 region after)
  ush* WAh   = (ush*)d_ws;                 // 8*65536        = 524288
  ush* XtT   = WAh   + 524288;             // 3200*256       = 819200
  ush* XtB   = XtT   + 819200;             // 12800*256      = 3276800
  ush* topP  = XtB   + 3276800;            // 3*16*1600*32   = 2457600
  ush* botP  = topP  + 2457600;            // 3*16*6400*32   = 9830400
  ush* topRb = botP  + 9830400;            // 16*1600*32     = 819200
  ush* botRb = topRb + 819200;             // 16*6400*32     = 3276800
  float* botR = (float*)(botRb + 3276800); // 16*6400*32 f32 = 3276800

  // 1) zero fold accumulator
  zero_kernel<<<1024, 256, 0, stream>>>(botR, 16u * NPIX_B * Dd);

  // 2) stage weights + activations as bf16 (fragment-ready layouts)
  prep_w<<<128, 256, 0, stream>>>(w_tk, w_tq, w_tv, w_bk, w_bq, w_bv,
                                  w_tf, w_bf, (unsigned*)WAh);
  prep_x<<<13, 256, 0, stream>>>(top, (unsigned*)XtT, NPIX_T);
  prep_x<<<50, 256, 0, stream>>>(bot, (unsigned*)XtB, NPIX_B);

  // 3) projections: waves = 3 * 16 Mtiles * (2*npix/64) Ngroups
  proj_gemm<<<600, 128, 0, stream>>>(XtT, WAh, (unsigned*)topP, NPIX_T, 50, 0);
  proj_gemm<<<2400, 128, 0, stream>>>(XtB, WAh, (unsigned*)botP, NPIX_B, 200, 3);

  // 4) attention + fold scatter: one wave per unit
  attn_kernel<<<UNITS / 4, 128, 0, stream>>>(topP, botP, topRb, botR);

  // 5) fold accumulator -> bf16
  conv_bf<<<6400, 256, 0, stream>>>(botR, (unsigned*)botRb, 16u * NPIX_B * Dd / 2);

  // 6) fuse convs -> outputs (top 819200 f32, then bot 3276800 f32)
  fuse_gemm<<<200, 128, 0, stream>>>(topRb, WAh, out, NPIX_T, 50, 6);
  fuse_gemm<<<800, 128, 0, stream>>>(botRb, WAh, out + 819200, NPIX_B, 200, 7);
}
